// VectorizedLIFLayer_38585986187646
// MI455X (gfx1250) — matrix operand order repaired and checked by reading, not compiled, not run on hardware
//
#include <hip/hip_runtime.h>

typedef _Float16 v16h __attribute__((ext_vector_type(16)));
typedef _Float16 v8h  __attribute__((ext_vector_type(8)));
typedef float    v8f  __attribute__((ext_vector_type(8)));

#define B_SZ   32
#define T_SZ   1024
#define D_SZ   512
#define H_SZ   512
#define HTILE  64     // per workgroup (2 waves x 32 columns)
#define TTILE  32     // timesteps per iteration (= WMMA M rows per wave)
#define WS_STRIDE 520 // padded halves per Ws row (1040 B: 16B-aligned, bank-spread)
#define AS_STRIDE 520
#define CS_STRIDE 33  // padded floats per Cs row

#define ALPHA_MEM 0.951229424500714f    // exp(-0.001/0.02)
#define ALPHA_SYN 0.8187307530779818f   // exp(-0.001/0.005)

static __device__ __forceinline__ v16h load_afrag(const _Float16* p) {
    // A 16x32 f16 in this code's k order (B uses the same): per lane one contiguous 16-half run
    v8h lo = *(const v8h*)(p);        // K = kk + half*16 + 0..7
    v8h hi = *(const v8h*)(p + 8);    // K = kk + half*16 + 8..15
    return __builtin_shufflevector(lo, hi, 0,1,2,3,4,5,6,7,8,9,10,11,12,13,14,15);
}

__global__ __launch_bounds__(64)
void lif_fused_wmma_kernel(const float* __restrict__ spikes,
                           const float* __restrict__ W,
                           const float* __restrict__ bias,
                           float* __restrict__ out)
{
    extern __shared__ char smem[];
    _Float16* Ws = (_Float16*)smem;                        // [HTILE][WS_STRIDE] f16, W^T tile
    _Float16* As = Ws + HTILE * WS_STRIDE;                 // [TTILE][AS_STRIDE] f16, spike tile
    float*    Cs = (float*)(As + TTILE * AS_STRIDE);       // [2][TTILE][CS_STRIDE] per-wave C

    const int tid  = threadIdx.x;
    const int wave = tid >> 5;
    const int lane = tid & 31;
    const int half = lane >> 4;
    const int ln   = lane & 15;

    const int h0 = blockIdx.x * HTILE;   // grid.x = 8
    const int b  = blockIdx.y;           // grid.y = 32
    const int hw = wave * 32;            // wave's 32 columns within the tile

    // ---- Stage W tile once: Ws[hl][k] = (f16) W[k][h0+hl]  (transposed) ----
    for (int idx = tid; idx < HTILE * D_SZ; idx += 64) {
        int hl = idx & (HTILE - 1);
        int k  = idx >> 6;
        Ws[hl * WS_STRIDE + k] = (_Float16)W[(size_t)k * H_SZ + h0 + hl];
    }

    const float bias_n = bias[h0 + hw + lane];
    float v_mem = 0.0f, i_syn = 0.0f;     // LIF state: one H column per lane
    float* cw = Cs + wave * TTILE * CS_STRIDE;

    for (int tb = 0; tb < T_SZ / TTILE; ++tb) {
        const int t0 = tb * TTILE;
        __syncthreads();   // Ws ready (iter 0) / previous tile's LDS reads done

        // ---- Stage A tile: As[m][k] = (f16) spikes[b][t0+m][k] ----
        for (int idx = tid; idx < TTILE * (D_SZ / 8); idx += 64) {
            int kc = idx & 63;          // 8-element chunk within row
            int m  = idx >> 6;          // timestep within tile
            const float4* src =
                (const float4*)(spikes + ((size_t)(b * T_SZ + t0 + m) * D_SZ) + kc * 8);
            float4 f0 = src[0];
            float4 f1 = src[1];
            v8h h8;
            h8[0] = (_Float16)f0.x; h8[1] = (_Float16)f0.y;
            h8[2] = (_Float16)f0.z; h8[3] = (_Float16)f0.w;
            h8[4] = (_Float16)f1.x; h8[5] = (_Float16)f1.y;
            h8[6] = (_Float16)f1.z; h8[7] = (_Float16)f1.w;
            *(v8h*)(As + m * AS_STRIDE + kc * 8) = h8;
            if (tb + 1 < T_SZ / TTILE)  // prefetch next time-tile (+TTILE rows)
                __builtin_prefetch(src + (size_t)TTILE * (D_SZ / 4), 0, 0);
        }
        __syncthreads();

        // ---- WMMA GEMM, 2x2 register blocking: 32t x 32h per wave, K = 512 ----
        v8f c00 = {0.f,0.f,0.f,0.f,0.f,0.f,0.f,0.f};
        v8f c01 = c00, c10 = c00, c11 = c00;
        const _Float16* a0p = As + ln * AS_STRIDE + half * 16;         // t rows 0..15
        const _Float16* a1p = As + (16 + ln) * AS_STRIDE + half * 16;  // t rows 16..31
        const _Float16* b0p = Ws + (hw + ln) * WS_STRIDE + half * 16;  // h cols 0..15
        const _Float16* b1p = Ws + (hw + 16 + ln) * WS_STRIDE + half * 16; // h 16..31
        #pragma unroll
        for (int kk = 0; kk < D_SZ; kk += 32) {
            v16h a0 = load_afrag(a0p + kk);
            v16h a1 = load_afrag(a1p + kk);
            v16h b0 = *(const v16h*)(b0p + kk);   // K = kk + half*16 + 0..15
            v16h b1 = *(const v16h*)(b1p + kk);
            c00 = __builtin_amdgcn_wmma_f32_16x16x32_f16(false, a0, false, b0, (short)0, c00, false, false);
            c01 = __builtin_amdgcn_wmma_f32_16x16x32_f16(false, a0, false, b1, (short)0, c01, false, false);
            c10 = __builtin_amdgcn_wmma_f32_16x16x32_f16(false, a1, false, b0, (short)0, c10, false, false);
            c11 = __builtin_amdgcn_wmma_f32_16x16x32_f16(false, a1, false, b1, (short)0, c11, false, false);
        }

        // ---- Spill C macro-tile (t-major) for the cross-lane time scan ----
        #pragma unroll
        for (int r = 0; r < 8; ++r) {
            const int row = r + 8 * half;
            cw[row * CS_STRIDE + ln]               = c00[r];
            cw[row * CS_STRIDE + 16 + ln]          = c01[r];
            cw[(16 + row) * CS_STRIDE + ln]        = c10[r];
            cw[(16 + row) * CS_STRIDE + 16 + ln]   = c11[r];
        }
        __syncthreads();

        // ---- Sequential LIF over 32 timesteps; every lane owns one H column ----
        float* orow = out + ((size_t)(b * T_SZ + t0) * H_SZ) + h0 + hw + lane;
        #pragma unroll
        for (int t = 0; t < TTILE; ++t) {
            float syn = cw[t * CS_STRIDE + lane] + bias_n;
            i_syn = ALPHA_SYN * i_syn + syn;
            v_mem = ALPHA_MEM * v_mem + i_syn;
            float o = (v_mem > 1.0f) ? 1.0f : 0.0f;
            orow[(size_t)t * H_SZ] = o;
            if (v_mem > 1.0f) v_mem = 0.0f;   // reset on spike
        }
    }
}

extern "C" void kernel_launch(void* const* d_in, const int* in_sizes, int n_in,
                              void* d_out, int out_size, void* d_ws, size_t ws_size,
                              hipStream_t stream) {
    (void)in_sizes; (void)n_in; (void)out_size; (void)d_ws; (void)ws_size;
    const float* spikes = (const float*)d_in[0];  // [32,1024,512]
    const float* W      = (const float*)d_in[1];  // [512,512]
    const float* bias   = (const float*)d_in[2];  // [512]
    float*       out    = (float*)d_out;          // [32,1024,512]

    dim3 grid(H_SZ / HTILE, B_SZ);   // (8, 32)
    dim3 block(64);                  // 2 wave32, 32t x 32h macro-tile each
    size_t shmem = (size_t)(HTILE * WS_STRIDE + TTILE * AS_STRIDE) * sizeof(_Float16)
                 + (size_t)2 * TTILE * CS_STRIDE * sizeof(float);   // ~106 KB
    lif_fused_wmma_kernel<<<grid, block, shmem, stream>>>(spikes, W, bias, out);
}
